// AAMSoftmax_normfree_39676907888375
// MI455X (gfx1250) — matrix operand order repaired and checked by reading, not compiled, not run on hardware
//
#include <hip/hip_runtime.h>

#define B_   1024
#define IN_  512
#define OUT_ 64000

typedef __attribute__((ext_vector_type(16))) _Float16 v16h;
typedef __attribute__((ext_vector_type(8)))  _Float16 v8h;
typedef __attribute__((ext_vector_type(8)))  float    v8f;
typedef __attribute__((ext_vector_type(4)))  int      v4i;

// ---- CDNA5 async global->LDS path (guarded; fallback = load + ds_store) ----
#if defined(__has_builtin)
#if __has_builtin(__builtin_amdgcn_global_load_async_to_lds_b128)
#define USE_ASYNC_LDS 1
#endif
#endif

#if defined(USE_ASYNC_LDS)
__device__ __forceinline__ void async_cp16(const void* g, void* l) {
    // builtin signature (from hipcc diagnostic): (v4i*, v4i*, imm offset, imm cpol)
    __builtin_amdgcn_global_load_async_to_lds_b128((v4i*)g, (v4i*)l, 0, 0);
}
#if __has_builtin(__builtin_amdgcn_s_wait_asynccnt)
#define WAIT_ASYNC(n) __builtin_amdgcn_s_wait_asynccnt(n)
#else
#define WAIT_ASYNC(n) asm volatile("s_wait_asynccnt %0" ::"i"(n))
#endif
#endif

// ---------------------------------------------------------------------------
// Kernel 0: scalar prep — cos(m), sin(m) once, not per GEMM thread.
// ---------------------------------------------------------------------------
__global__ void prep_trig_kernel(const float* __restrict__ m, float* __restrict__ trig) {
    if (threadIdx.x == 0) {
        const float v = m[0];
        trig[0] = cosf(v);
        trig[1] = sinf(v);
    }
}

// ---------------------------------------------------------------------------
// Kernel 1: per-row normalize x (one wave32 per row).
// ---------------------------------------------------------------------------
__global__ __launch_bounds__(128) void norm_x_kernel(const float* __restrict__ x,
                                                     _Float16* __restrict__ xnf,
                                                     float* __restrict__ norms) {
    const int wave = threadIdx.x >> 5;
    const int lane = threadIdx.x & 31;
    const int row  = blockIdx.x * 4 + wave;

    const float* xr = x + (size_t)row * IN_;
    float v[16];
    float ss = 0.f;
#pragma unroll
    for (int i = 0; i < 16; ++i) {
        v[i] = xr[lane + 32 * i];
        ss += v[i] * v[i];
    }
#pragma unroll
    for (int off = 16; off >= 1; off >>= 1)
        ss += __shfl_xor(ss, off, 32);

    const float nrm = sqrtf(ss);
    const float inv = 1.f / fmaxf(nrm, 1e-12f);

    _Float16* xo = xnf + (size_t)row * IN_;
#pragma unroll
    for (int i = 0; i < 16; ++i)
        xo[lane + 32 * i] = (_Float16)(v[i] * inv);

    if (lane == 0) norms[row] = nrm;
}

// ---------------------------------------------------------------------------
// Kernel 2: per-row normalize weight; writes f32 wn output + f16 GEMM copy.
// ---------------------------------------------------------------------------
__global__ __launch_bounds__(256) void norm_w_kernel(const float* __restrict__ w,
                                                     _Float16* __restrict__ wnf,
                                                     float* __restrict__ wn32) {
    const int wave = threadIdx.x >> 5;
    const int lane = threadIdx.x & 31;
    const int row  = blockIdx.x * 8 + wave;

    const float* wr = w + (size_t)row * IN_;
    float v[16];
    float ss = 0.f;
#pragma unroll
    for (int i = 0; i < 16; ++i) {
        v[i] = wr[lane + 32 * i];
        ss += v[i] * v[i];
    }
#pragma unroll
    for (int off = 16; off >= 1; off >>= 1)
        ss += __shfl_xor(ss, off, 32);

    const float inv = 1.f / fmaxf(sqrtf(ss), 1e-12f);

    float*    o32 = wn32 + (size_t)row * IN_;
    _Float16* o16 = wnf  + (size_t)row * IN_;
#pragma unroll
    for (int i = 0; i < 16; ++i) {
        const float t = v[i] * inv;
        o32[lane + 32 * i] = t;
        o16[lane + 32 * i] = (_Float16)t;
    }
}

// ---------------------------------------------------------------------------
// Kernel 3: WMMA GEMM cosine = xn @ wn^T (M=1024, N=64000, K=512)
// + fused AAM margin epilogue + norm scaling.
// 256 threads (8 wave32), 128x128 tile, K-step 32, double-buffered LDS with
// async global->LDS copies (ASYNCcnt) when available.
// ---------------------------------------------------------------------------
__global__ __launch_bounds__(256) void aam_gemm_kernel(
    const _Float16* __restrict__ xnf, const _Float16* __restrict__ wnf,
    const float* __restrict__ norms, const int* __restrict__ label,
    const float* __restrict__ trig, float* __restrict__ out) {

    __shared__ __align__(16) _Float16 lA[2][128][40];  // 2 x 10 KB
    __shared__ __align__(16) _Float16 lB[2][128][40];

    const int tid    = threadIdx.x;
    const int lane   = tid & 31;
    const int waveId = tid >> 5;
    const int waveM  = waveId >> 2;  // 0..1 -> 64-row slab
    const int waveN  = waveId & 3;   // 0..3 -> 32-col slab
    const int bM = blockIdx.y * 128;
    const int bN = blockIdx.x * 128;

    // cooperative tile-load mapping: 256 threads x 32B
    const int ldr = tid >> 1;
    const int ldc = (tid & 1) * 16;
    const _Float16* gAbase = xnf + (size_t)(bM + ldr) * IN_ + ldc;
    const _Float16* gBbase = wnf + (size_t)(bN + ldr) * IN_ + ldc;

    // wave32 fragment layouts (ISA 7.12.2):
    const int mrow  = lane & 15;
    const int kOffA = (lane >> 4) * 8;
    const int kOffB = (lane >> 4) * 8;

    v8f acc[4][2];
#pragma unroll
    for (int mt = 0; mt < 4; ++mt)
#pragma unroll
        for (int nt = 0; nt < 2; ++nt)
            acc[mt][nt] = {};

    auto issue = [&](int kk, int buf) {
        const _Float16* gA = gAbase + kk * 32;
        const _Float16* gB = gBbase + kk * 32;
#if defined(USE_ASYNC_LDS)
        async_cp16(gA,     &lA[buf][ldr][ldc]);
        async_cp16(gA + 8, &lA[buf][ldr][ldc + 8]);
        async_cp16(gB,     &lB[buf][ldr][ldc]);
        async_cp16(gB + 8, &lB[buf][ldr][ldc + 8]);
#else
        v8h a0 = *(const v8h*)gA;
        v8h a1 = *(const v8h*)(gA + 8);
        v8h b0 = *(const v8h*)gB;
        v8h b1 = *(const v8h*)(gB + 8);
        *(v8h*)&lA[buf][ldr][ldc]     = a0;
        *(v8h*)&lA[buf][ldr][ldc + 8] = a1;
        *(v8h*)&lB[buf][ldr][ldc]     = b0;
        *(v8h*)&lB[buf][ldr][ldc + 8] = b1;
#endif
        // prefetch the K-chunk after next into L2/L0
        __builtin_prefetch(gA + 64, 0, 0);
        __builtin_prefetch(gB + 64, 0, 0);
    };

    constexpr int KSTEPS = IN_ / 32;  // 16
    int buf = 0;
    issue(0, 0);

    for (int kk = 0; kk < KSTEPS; ++kk) {
        if (kk + 1 < KSTEPS) {
            issue(kk + 1, buf ^ 1);
#if defined(USE_ASYNC_LDS)
            WAIT_ASYNC(4);  // previous tile's 4 async ops have landed
#endif
        } else {
#if defined(USE_ASYNC_LDS)
            WAIT_ASYNC(0);
#endif
        }
        __syncthreads();  // tile[buf] visible to all waves

        v16h afrag[4], bfrag[2];
#pragma unroll
        for (int mt = 0; mt < 4; ++mt) {
            const int r = waveM * 64 + mt * 16 + mrow;
            v8h lo = *(const v8h*)&lA[buf][r][kOffA];
            v8h hi = *(const v8h*)&lA[buf][r][16 + kOffA];
#pragma unroll
            for (int i = 0; i < 8; ++i) {
                afrag[mt][i]     = lo[i];
                afrag[mt][i + 8] = hi[i];
            }
        }
#pragma unroll
        for (int nt = 0; nt < 2; ++nt) {
            const int c = waveN * 32 + nt * 16 + mrow;
            v8h lo = *(const v8h*)&lB[buf][c][kOffB];
            v8h hi = *(const v8h*)&lB[buf][c][16 + kOffB];
#pragma unroll
            for (int i = 0; i < 8; ++i) {
                bfrag[nt][i]     = lo[i];
                bfrag[nt][i + 8] = hi[i];
            }
        }

#pragma unroll
        for (int mt = 0; mt < 4; ++mt)
#pragma unroll
            for (int nt = 0; nt < 2; ++nt)
                acc[mt][nt] = __builtin_amdgcn_wmma_f32_16x16x32_f16(
                    false, afrag[mt], false, bfrag[nt],
                    (short)0, acc[mt][nt], false, false);

        __syncthreads();  // everyone done reading tile[buf] before it's rewritten
        buf ^= 1;
    }

    // fused epilogue: margin substitution at (row, label[row]) when cos>0,
    // then scale by raw ||x||.  C layout: VGPR j -> M=j (lanes<16) or j+8,
    // N = lane%16.
    const float cosm = trig[0];
    const float sinm = trig[1];

#pragma unroll
    for (int mt = 0; mt < 4; ++mt) {
        const int rbase = bM + waveM * 64 + mt * 16 + ((lane >> 4) << 3);
        float nrm8[8];
        int   lbl8[8];
#pragma unroll
        for (int j = 0; j < 8; ++j) {
            nrm8[j] = norms[rbase + j];
            lbl8[j] = label[rbase + j];
        }
#pragma unroll
        for (int nt = 0; nt < 2; ++nt) {
            const int col = bN + waveN * 32 + nt * 16 + (lane & 15);
#pragma unroll
            for (int j = 0; j < 8; ++j) {
                const float c = acc[mt][nt][j];
                float val = c;
                if (lbl8[j] == col && c > 0.f) {
                    const float s2 = fmaxf(1.f - c * c, 0.f);
                    val = c * cosm - sqrtf(s2) * sinm;
                }
                out[(size_t)(rbase + j) * OUT_ + col] = nrm8[j] * val;
            }
        }
    }
}

// ---------------------------------------------------------------------------
extern "C" void kernel_launch(void* const* d_in, const int* in_sizes, int n_in,
                              void* d_out, int out_size, void* d_ws, size_t ws_size,
                              hipStream_t stream) {
    const float* x      = (const float*)d_in[0];
    const int*   label  = (const int*)d_in[1];
    const float* weight = (const float*)d_in[2];
    // d_in[3] = s (unused by the normfree output)
    const float* m      = (const float*)d_in[4];

    float* out_logits = (float*)d_out;                       // [B, OUT]
    float* out_wn     = (float*)d_out + (size_t)B_ * OUT_;   // [OUT, IN]

    // workspace: norms (4KB) | trig (2 f32, padded) | xn f16 (1MB) | wn f16 (64MB)
    char*     ws    = (char*)d_ws;
    float*    norms = (float*)ws;
    float*    trig  = (float*)(ws + 4096);
    _Float16* xnf   = (_Float16*)(ws + 8192);
    _Float16* wnf   = (_Float16*)(ws + 8192 + (size_t)B_ * IN_ * sizeof(_Float16));

    prep_trig_kernel<<<1, 32, 0, stream>>>(m, trig);
    norm_x_kernel<<<B_ / 4, 128, 0, stream>>>(x, xnf, norms);
    norm_w_kernel<<<OUT_ / 8, 256, 0, stream>>>(weight, wnf, out_wn);

    dim3 grid(OUT_ / 128, B_ / 128);  // 500 x 8
    aam_gemm_kernel<<<grid, 256, 0, stream>>>(xnf, wnf, norms, label, trig, out_logits);
}
